// GatedTestLayer_32126355374902
// MI455X (gfx1250) — matrix operand order repaired and checked by reading, not compiled, not run on hardware
//
#include <hip/hip_runtime.h>
#include <hip/hip_bf16.h>

#define NN 50000
#define NE 1600000
#define DD 64
#define ND ((size_t)NN * DD)       // 3,200,000
#define ED ((size_t)NE * DD)       // 102,400,000
#define BN_EPS 1e-5f

typedef __bf16 bf16;
typedef __attribute__((ext_vector_type(16))) bf16 v16bf;
typedef __attribute__((ext_vector_type(8)))  float v8f;
typedef __attribute__((ext_vector_type(4)))  unsigned int u32x4;
typedef __attribute__((ext_vector_type(8)))  int i32x8;
typedef __attribute__((ext_vector_type(4)))  int i32x4;

#define WMMA_BF16(a, b, c) \
    __builtin_amdgcn_wmma_f32_16x16x32_bf16(false, (a), false, (b), (short)0, (c), false, false)

// ---------------------------------------------------------------------------
// TDM: DMA a rows x cols f32 tile (tile == tensor, stride = cols) into LDS.
// D# built per CDNA5 ISA 8.3/8.4: group0 = {count=1, lds_addr, global_addr,
// type=2}; group1 = {data_size=4B, tensor_dim0/1, tile_dim0/1, dim0_stride}.
// This toolchain exposes the 6-arg builtin (g2, g3, extra i32x8 group, cpol).
// Issue from one wave, wait TENSORcnt, then __syncthreads in caller.
// ---------------------------------------------------------------------------
__device__ __forceinline__ void tdm_load_2d_f32(unsigned lds_off, const void* gaddr,
                                                unsigned rows, unsigned cols) {
    unsigned long long ga = (unsigned long long)(uintptr_t)gaddr;
    u32x4 g0;
    g0[0] = 1u;                                            // count=1, user mode
    g0[1] = lds_off;                                       // lds_addr (bytes)
    g0[2] = (unsigned)(ga & 0xFFFFFFFFu);                  // global_addr[31:0]
    g0[3] = (unsigned)((ga >> 32) & 0x01FFFFFFu) | (2u << 30); // addr[56:32], type=2
    i32x8 g1;
    g1[0] = (int)(2u << 16);                               // wg_mask=0, data_size=2 (4B)
    g1[1] = (int)((cols & 0xFFFFu) << 16);                 // tensor_dim0[15:0]
    g1[2] = (int)(((cols >> 16) & 0xFFFFu) | ((rows & 0xFFFFu) << 16)); // dim0 hi | dim1 lo
    g1[3] = (int)(((rows >> 16) & 0xFFFFu) | ((cols & 0xFFFFu) << 16)); // dim1 hi | tile_dim0
    g1[4] = (int)(rows & 0xFFFFu);                         // tile_dim1 | tile_dim2=0
    g1[5] = (int)cols;                                     // tensor_dim0_stride[31:0]
    g1[6] = 0;                                             // stride hi | dim1_stride lo
    g1[7] = 0;
    i32x4 gz4 = {0, 0, 0, 0};
    i32x8 gz8 = {0, 0, 0, 0, 0, 0, 0, 0};
    __builtin_amdgcn_tensor_load_to_lds(g0, g1, gz4, gz4, gz8, 0);
    __builtin_amdgcn_s_wait_tensorcnt(0);
}

// ---------------------------------------------------------------------------
// Fragment helpers (wave32 layouts per CDNA5 ISA 7.12.2)
// A 16x32 bf16: lane<16 holds K{0..7,16..23} of row M=lane; lane>=16 holds
//   K{8..15,24..31} of row M=lane-16.  -> two float4 pairs, convert to bf16.
// B 32x16 bf16: lane holds column N=n0+(lane&15); lanes<16 cover K{0..7,16..23},
//   lanes>=16 cover K{8..15,24..31}.  Staged fragment-contiguous in LDS (32B/lane).
// C/D v8f: element j is (M = j + 8*(lane>>4), N = n0 + (lane&15)).
// ---------------------------------------------------------------------------
__device__ __forceinline__ v16bf load_A_frag_f32(const float* __restrict__ rows,
                                                 int lane, int kbase) {
    int m  = lane & 15;
    int kk = (lane >> 4) << 3;          // 0 or 8
    const float* rp = rows + (size_t)m * DD + kbase + kk;
    float4 x0 = ((const float4*)rp)[0];
    float4 x1 = ((const float4*)rp)[1];
    float4 y0 = ((const float4*)(rp + 16))[0];
    float4 y1 = ((const float4*)(rp + 16))[1];
    v16bf a;
    a[0]  = (bf16)x0.x; a[1]  = (bf16)x0.y; a[2]  = (bf16)x0.z; a[3]  = (bf16)x0.w;
    a[4]  = (bf16)x1.x; a[5]  = (bf16)x1.y; a[6]  = (bf16)x1.z; a[7]  = (bf16)x1.w;
    a[8]  = (bf16)y0.x; a[9]  = (bf16)y0.y; a[10] = (bf16)y0.z; a[11] = (bf16)y0.w;
    a[12] = (bf16)y1.x; a[13] = (bf16)y1.y; a[14] = (bf16)y1.z; a[15] = (bf16)y1.w;
    return a;
}

// Convert raw f32 weight (LDS) into fragment-contiguous bf16 layout (LDS).
// Wfrag index = frag*512 + lane*16 + i ; frag = ks*4 + t.
__device__ __forceinline__ void stage_W_frags(const float* __restrict__ Wraw,
                                              bf16* __restrict__ Wfrag, int nfrag) {
    for (int idx = threadIdx.x; idx < nfrag * 512; idx += blockDim.x) {
        int i  = idx & 15;
        int ln = (idx >> 4) & 31;
        int fr = idx >> 9;
        int ks = fr >> 2, t = fr & 3;
        int n  = 16 * t + (ln & 15);
        int k  = 32 * ks + ((ln >> 4) << 3) + (i & 7) + ((i >> 3) << 4);
        Wfrag[idx] = (bf16)Wraw[k * DD + n];
    }
}

__device__ __forceinline__ v16bf load_B_frag(const bf16* __restrict__ Wfrag,
                                             int lane, int frag) {
    return *(const v16bf*)&Wfrag[(frag * 32 + lane) * 16];   // 32B -> 2x ds_load_b128
}

// ---------------------------------------------------------------------------
// K0: zero scratch (sum_sigma, fsum, stats, bn)
// ---------------------------------------------------------------------------
__global__ void zero_kernel(float* __restrict__ p, size_t n) {
    size_t i = (size_t)blockIdx.x * blockDim.x + threadIdx.x;
    size_t s = (size_t)gridDim.x * blockDim.x;
    for (; i < n; i += s) p[i] = 0.0f;
}

// ---------------------------------------------------------------------------
// K1: out = h @ Wsel + bsel.  blockIdx.y selects {A,B,D,E}.  B-fragments
// hoisted into registers; W brought in by TDM; h is L2-resident across the
// 4 y-slices.
// ---------------------------------------------------------------------------
__global__ void node_gemm_kernel(const float* __restrict__ h,
                                 const float* __restrict__ WA, const float* __restrict__ bA,
                                 const float* __restrict__ WB, const float* __restrict__ bB,
                                 const float* __restrict__ WD, const float* __restrict__ bD,
                                 const float* __restrict__ WE, const float* __restrict__ bE,
                                 float* __restrict__ Ah, float* __restrict__ Bh,
                                 float* __restrict__ Dh, float* __restrict__ Eh) {
    __shared__ __align__(16) float Wraw[DD * DD];       // 16 KB
    __shared__ __align__(32) bf16  Wfrag[8 * 512];      // 8 KB
    const int wy = blockIdx.y;
    const float* Wsel = (wy == 0) ? WA : (wy == 1) ? WB : (wy == 2) ? WD : WE;
    const float* bsel = (wy == 0) ? bA : (wy == 1) ? bB : (wy == 2) ? bD : bE;
    float*       osel = (wy == 0) ? Ah : (wy == 1) ? Bh : (wy == 2) ? Dh : Eh;

    if (threadIdx.x < 32)
        tdm_load_2d_f32((unsigned)(uintptr_t)&Wraw[0], Wsel, DD, DD);
    __syncthreads();
    stage_W_frags(Wraw, Wfrag, 8);
    __syncthreads();

    const int lane = threadIdx.x & 31;
    const int hi   = lane >> 4;
    const int nl   = lane & 15;

    v16bf Bfr[2][4];
#pragma unroll
    for (int ks = 0; ks < 2; ++ks)
#pragma unroll
        for (int t = 0; t < 4; ++t)
            Bfr[ks][t] = load_B_frag(Wfrag, lane, ks * 4 + t);

    float bb[4];
#pragma unroll
    for (int t = 0; t < 4; ++t) bb[t] = bsel[16 * t + nl];

    const int nwaves = (gridDim.x * blockDim.x) >> 5;
    const int wave   = (blockIdx.x * blockDim.x + threadIdx.x) >> 5;
    const int ntiles = NN / 16;         // 3125

    for (int tile = wave; tile < ntiles; tile += nwaves) {
        const int r0 = tile * 16;
        v16bf a0 = load_A_frag_f32(h + (size_t)r0 * DD, lane, 0);
        v16bf a1 = load_A_frag_f32(h + (size_t)r0 * DD, lane, 32);
#pragma unroll
        for (int t = 0; t < 4; ++t) {
            v8f acc = {};
            acc = WMMA_BF16(a0, Bfr[0][t], acc);
            acc = WMMA_BF16(a1, Bfr[1][t], acc);
            const int col = 16 * t + nl;
#pragma unroll
            for (int j = 0; j < 8; ++j)
                osel[(size_t)(r0 + j + 8 * hi) * DD + col] = acc[j] + bb[t];
        }
    }
}

// ---------------------------------------------------------------------------
// K2: fused edge pass: e_new = e@WC + bC + Dh[src] + Eh[dst]   (WMMA)
//     sigma = sigmoid(e_new); atomic scatter-add into sum_sigma[dst]
//     + accumulate e-BatchNorm column sum/sumsq (regs -> LDS -> global)
// ---------------------------------------------------------------------------
__global__ void edge_gemm_gate_kernel(const float* __restrict__ e,
                                      const int* __restrict__ src,
                                      const int* __restrict__ dst,
                                      const float* __restrict__ WC,
                                      const float* __restrict__ bC,
                                      const float* __restrict__ Dh,
                                      const float* __restrict__ Eh,
                                      float* __restrict__ e_new,
                                      float* __restrict__ sum_sigma,
                                      float* __restrict__ e_stats) {
    __shared__ __align__(16) float Wraw[DD * DD];       // 16 KB
    __shared__ __align__(32) bf16  Wfrag[8 * 512];      // 8 KB
    __shared__ float lds_stats[2 * DD];

    if (threadIdx.x < 32)
        tdm_load_2d_f32((unsigned)(uintptr_t)&Wraw[0], WC, DD, DD);
    for (int i = threadIdx.x; i < 2 * DD; i += blockDim.x) lds_stats[i] = 0.0f;
    __syncthreads();
    stage_W_frags(Wraw, Wfrag, 8);
    __syncthreads();

    const int lane = threadIdx.x & 31;
    const int hi   = lane >> 4;
    const int nl   = lane & 15;

    v16bf Bfr[2][4];
#pragma unroll
    for (int ks = 0; ks < 2; ++ks)
#pragma unroll
        for (int t = 0; t < 4; ++t)
            Bfr[ks][t] = load_B_frag(Wfrag, lane, ks * 4 + t);

    float bb[4];
#pragma unroll
    for (int t = 0; t < 4; ++t) bb[t] = bC[16 * t + nl];

    const int nwaves = (gridDim.x * blockDim.x) >> 5;
    const int wave   = (blockIdx.x * blockDim.x + threadIdx.x) >> 5;
    const int ntiles = NE / 16;         // 100000

    float st_sum[4] = {0.f, 0.f, 0.f, 0.f};
    float st_sq[4]  = {0.f, 0.f, 0.f, 0.f};

    for (int tile = wave; tile < ntiles; tile += nwaves) {
        const int r0 = tile * 16;
        if (tile + nwaves < ntiles)     // stream hint -> global_prefetch_b8
            __builtin_prefetch(e + (size_t)(tile + nwaves) * 16 * DD, 0, 3);

        v16bf a0 = load_A_frag_f32(e + (size_t)r0 * DD, lane, 0);
        v16bf a1 = load_A_frag_f32(e + (size_t)r0 * DD, lane, 32);

        int s8[8], d8[8];
#pragma unroll
        for (int j = 0; j < 8; ++j) {
            int row = r0 + 8 * hi + j;
            s8[j] = src[row];
            d8[j] = dst[row];
        }
#pragma unroll
        for (int t = 0; t < 4; ++t) {
            v8f acc = {};
            acc = WMMA_BF16(a0, Bfr[0][t], acc);
            acc = WMMA_BF16(a1, Bfr[1][t], acc);
            const int col = 16 * t + nl;
#pragma unroll
            for (int j = 0; j < 8; ++j) {
                float v = acc[j] + bb[t] + Dh[(size_t)s8[j] * DD + col]
                                         + Eh[(size_t)d8[j] * DD + col];
                e_new[(size_t)(r0 + j + 8 * hi) * DD + col] = v;
                float sg = 1.0f / (1.0f + __expf(-v));
                atomicAdd(&sum_sigma[(size_t)d8[j] * DD + col], sg);
                st_sum[t] += v;
                st_sq[t]  += v * v;
            }
        }
    }
#pragma unroll
    for (int t = 0; t < 4; ++t) {
        int col = 16 * t + nl;
        atomicAdd(&lds_stats[col],      st_sum[t]);
        atomicAdd(&lds_stats[DD + col], st_sq[t]);
    }
    __syncthreads();
    for (int i = threadIdx.x; i < 2 * DD; i += blockDim.x)
        atomicAdd(&e_stats[i], lds_stats[i]);
}

// ---------------------------------------------------------------------------
// K3: eee = Bh / (sum_sigma + 1e-6)
// ---------------------------------------------------------------------------
__global__ void node_eee_kernel(const float* __restrict__ Bh,
                                const float* __restrict__ sum_sigma,
                                float* __restrict__ eee) {
    size_t i = (size_t)blockIdx.x * blockDim.x + threadIdx.x;
    size_t s = (size_t)gridDim.x * blockDim.x;
    for (; i < ND; i += s) eee[i] = Bh[i] / (sum_sigma[i] + 1e-6f);
}

// ---------------------------------------------------------------------------
// K4: generic BN stat finalize: mu/rstd from sum/sumsq
// ---------------------------------------------------------------------------
__global__ void bn_finalize_kernel(const float* __restrict__ stats, float n,
                                   float* __restrict__ mu_out,
                                   float* __restrict__ rstd_out) {
    int i = threadIdx.x;
    if (i < DD) {
        float mu  = stats[i] / n;
        float var = stats[DD + i] / n - mu * mu;
        mu_out[i]   = mu;
        rstd_out[i] = rsqrtf(var + BN_EPS);
    }
}

// ---------------------------------------------------------------------------
// K5: fused edge message + e BN apply.  Reads e_new once: sigma from the
// pre-BN value feeds tanh-message scatter into fsum[dst]; the normalized
// ReLU'd value is written back in place (saves a full 410MB pass).
// Stride is a multiple of 64 so each thread owns a fixed column.
// ---------------------------------------------------------------------------
__global__ void edge_msg_bn_kernel(float* __restrict__ e_io,
                                   const int* __restrict__ src,
                                   const int* __restrict__ dst,
                                   const float* __restrict__ eee,
                                   const float* __restrict__ w_param,
                                   const float* __restrict__ b_param,
                                   const float* __restrict__ e_mu,
                                   const float* __restrict__ e_rstd,
                                   const float* __restrict__ gamma_e,
                                   const float* __restrict__ beta_e,
                                   float* __restrict__ fsum) {
    size_t idx = (size_t)blockIdx.x * blockDim.x + threadIdx.x;
    size_t s   = (size_t)gridDim.x * blockDim.x;   // multiple of 64
    const int col = (int)(idx & 63);
    const float w  = __expf(w_param[col]);
    const float b  = b_param[col];
    const float mu = e_mu[col];
    const float rs = e_rstd[col];
    const float g  = gamma_e[col];
    const float bt = beta_e[col];
    for (size_t i = idx; i < ED; i += s) {
        size_t edge = i >> 6;
        float v  = e_io[i];
        float sg = 1.0f / (1.0f + __expf(-v));
        float m  = eee[(size_t)src[edge] * DD + col] * sg;
        float t  = tanhf(w * m + b);
        atomicAdd(&fsum[(size_t)dst[edge] * DD + col], t);
        float o = (v - mu) * rs * g + bt;
        e_io[i] = fmaxf(o, 0.0f);
    }
}

// ---------------------------------------------------------------------------
// K6: h_pre = Ah + (atanh(clip(fsum)) - b)/w, accumulate h-BN stats
// ---------------------------------------------------------------------------
__global__ void node_finish_kernel(const float* __restrict__ Ah,
                                   const float* __restrict__ fsum,
                                   const float* __restrict__ w_param,
                                   const float* __restrict__ b_param,
                                   float* __restrict__ h_pre,
                                   float* __restrict__ h_stats) {
    __shared__ float lds_stats[2 * DD];
    for (int i = threadIdx.x; i < 2 * DD; i += blockDim.x) lds_stats[i] = 0.0f;
    __syncthreads();

    size_t idx = (size_t)blockIdx.x * blockDim.x + threadIdx.x;
    size_t s   = (size_t)gridDim.x * blockDim.x;   // multiple of 64
    const int col = (int)(idx & 63);
    const float w = __expf(w_param[col]);
    const float b = b_param[col];
    float acc = 0.f, accsq = 0.f;
    for (size_t i = idx; i < ND; i += s) {
        float f = fsum[i];
        f = fminf(0.99f, fmaxf(-0.99f, f));
        float at = 0.5f * __logf((1.0f + f) / (1.0f - f));
        float v  = Ah[i] + (at - b) / w;
        h_pre[i] = v;
        acc   += v;
        accsq += v * v;
    }
    atomicAdd(&lds_stats[col],      acc);
    atomicAdd(&lds_stats[DD + col], accsq);
    __syncthreads();
    for (int i = threadIdx.x; i < 2 * DD; i += blockDim.x)
        atomicAdd(&h_stats[i], lds_stats[i]);
}

// ---------------------------------------------------------------------------
// K7: h BN apply + ReLU (in place)
// ---------------------------------------------------------------------------
__global__ void bn_apply_kernel(float* __restrict__ x, size_t n,
                                const float* __restrict__ mu_v,
                                const float* __restrict__ rstd_v,
                                const float* __restrict__ gamma,
                                const float* __restrict__ beta) {
    size_t idx = (size_t)blockIdx.x * blockDim.x + threadIdx.x;
    size_t s   = (size_t)gridDim.x * blockDim.x;   // multiple of 64
    const int col = (int)(idx & 63);
    const float mu = mu_v[col];
    const float rs = rstd_v[col];
    const float g  = gamma[col];
    const float bt = beta[col];
    for (size_t i = idx; i < n; i += s) {
        float v = (x[i] - mu) * rs * g + bt;
        x[i] = fmaxf(v, 0.0f);
    }
}

// ---------------------------------------------------------------------------
extern "C" void kernel_launch(void* const* d_in, const int* in_sizes, int n_in,
                              void* d_out, int out_size, void* d_ws, size_t ws_size,
                              hipStream_t stream) {
    const float* h       = (const float*)d_in[0];
    const float* e       = (const float*)d_in[1];
    const int*   src     = (const int*)  d_in[2];
    const int*   dst     = (const int*)  d_in[3];
    const float* WA      = (const float*)d_in[4];
    const float* bA      = (const float*)d_in[5];
    const float* WB      = (const float*)d_in[6];
    const float* bB      = (const float*)d_in[7];
    const float* WC      = (const float*)d_in[8];
    const float* bC      = (const float*)d_in[9];
    const float* WD      = (const float*)d_in[10];
    const float* bD      = (const float*)d_in[11];
    const float* WE      = (const float*)d_in[12];
    const float* bE      = (const float*)d_in[13];
    const float* w_param = (const float*)d_in[14];
    const float* b_param = (const float*)d_in[15];
    const float* gamma_h = (const float*)d_in[16];
    const float* beta_h  = (const float*)d_in[17];
    const float* gamma_e = (const float*)d_in[18];
    const float* beta_e  = (const float*)d_in[19];

    float* h_out = (float*)d_out;            // [NN*DD]
    float* e_out = (float*)d_out + ND;       // [NE*DD] (e_new pre-BN, then in-place)

    float* ws        = (float*)d_ws;
    float* Ah        = ws + 0 * ND;
    float* Bh        = ws + 1 * ND;
    float* Dh        = ws + 2 * ND;
    float* Eh        = ws + 3 * ND;
    float* eee       = ws + 4 * ND;
    float* sum_sigma = ws + 5 * ND;
    float* fsum      = ws + 6 * ND;
    float* h_stats   = ws + 7 * ND;          // 128 floats
    float* e_stats   = h_stats + 2 * DD;     // 128 floats
    float* bn        = e_stats + 2 * DD;     // 256: h_mu,h_rstd,e_mu,e_rstd

    // zero: sum_sigma, fsum, h_stats, e_stats, bn
    zero_kernel<<<1024, 256, 0, stream>>>(sum_sigma, 2 * ND + 8 * DD);

    node_gemm_kernel<<<dim3(64, 4), 256, 0, stream>>>(
        h, WA, bA, WB, bB, WD, bD, WE, bE, Ah, Bh, Dh, Eh);

    edge_gemm_gate_kernel<<<512, 256, 0, stream>>>(e, src, dst, WC, bC, Dh, Eh,
                                                   e_out, sum_sigma, e_stats);

    node_eee_kernel<<<512, 256, 0, stream>>>(Bh, sum_sigma, eee);

    bn_finalize_kernel<<<1, 64, 0, stream>>>(e_stats, (float)NE,
                                             bn + 2 * DD, bn + 3 * DD);

    edge_msg_bn_kernel<<<2048, 256, 0, stream>>>(e_out, src, dst, eee,
                                                 w_param, b_param,
                                                 bn + 2 * DD, bn + 3 * DD,
                                                 gamma_e, beta_e, fsum);

    node_finish_kernel<<<256, 256, 0, stream>>>(Ah, fsum, w_param, b_param,
                                                h_out, h_stats);

    bn_finalize_kernel<<<1, 64, 0, stream>>>(h_stats, (float)NN, bn, bn + DD);

    bn_apply_kernel<<<256, 256, 0, stream>>>(h_out, ND, bn, bn + DD,
                                             gamma_h, beta_h);
}
